// PatchSVAEv2_49976239456538
// MI455X (gfx1250) — matrix operand order repaired and checked by reading, not compiled, not run on hardware
//
#include <hip/hip_runtime.h>
#include <math.h>

typedef __attribute__((ext_vector_type(16))) _Float16 v16h;
typedef __attribute__((ext_vector_type(8)))  _Float16 v8h;
typedef __attribute__((ext_vector_type(4)))  _Float16 v4h;
typedef __attribute__((ext_vector_type(8)))  float    v8f;
typedef __attribute__((ext_vector_type(4)))  float    f4;

__device__ __forceinline__ float gelu_f(float x) {
    return 0.5f * x * (1.0f + erff(x * 0.70710678118654752f));
}

// ---------------------------------------------------------------------------
// Fused GEMM:  C = act(A @ W + bias) (+ resid)
// A: MxK fp32 row-major, W: KxN fp32 row-major, C: MxN fp32.
// f16 WMMA 16x16x32, fp32 accumulate. Block tile 64x128, 8 waves (wave32).
// LDS tiles are stored pre-swizzled into WMMA fragment order so each lane's
// fragment is 32 contiguous bytes -> 2x ds_load_b128 per fragment.
// Double-buffered LDS: one barrier per K step, global loads overlap WMMA.
// All B fragments are preloaded before the WMMA burst so DS latency overlaps
// matrix-pipe execution instead of serializing load->wait->wmma.
// ---------------------------------------------------------------------------
#define FR 24   // halves per fragment record (16 used + 8 pad; 48B, 16B-aligned)

__global__ __launch_bounds__(256) void k_gemm_wmma(
    const float* __restrict__ A, const float* __restrict__ W,
    const float* __restrict__ bias, const float* __restrict__ resid,
    float* __restrict__ C, int M, int K, int N, int act)
{
  // aS: [buf][(mTile*32 + lane)*FR + e]   4 mTiles  -> 4*32*FR halves
  // bS: [buf][(nTile*32 + lane)*FR + e]   8 nTiles  -> 8*32*FR halves
  __shared__ _Float16 aS[2][4 * 32 * FR];
  __shared__ _Float16 bS[2][8 * 32 * FR];

  const int tid  = threadIdx.x;
  const int wid  = tid >> 5;
  const int lane = tid & 31;
  const int m0 = blockIdx.y * 64;
  const int n0 = blockIdx.x * 128;
  const int waveM = wid & 3;   // 16-row strip
  const int waveN = wid >> 2;  // 64-col strip
  const int row_l = lane & 15;
  const int kgrp  = lane >> 4;

  union Frag { v16h v; v8h h[2]; };

  v8f acc[4];
  for (int i = 0; i < 4; ++i)
    for (int j = 0; j < 8; ++j) acc[i][j] = 0.0f;

  // stage one 64x32 A tile (fp32 -> f16), swizzled to fragment order
  auto stageA = [&](int buf, int k0) {
#pragma unroll
    for (int j = 0; j < 2; ++j) {
      int task = tid + 256 * j;       // 512 quad-tasks
      int r    = task >> 3;           // 0..63
      int kk4  = (task & 7) * 4;      // 0,4,...,28
      int gm = m0 + r, gk = k0 + kk4;
      f4 vf = {0.0f, 0.0f, 0.0f, 0.0f};
      if (gm < M && gk < K) vf = *(const f4*)&A[(size_t)gm * K + gk];  // K%16==0
      int kg = (kk4 >> 3) & 1;
      int e  = (kk4 & 7) + ((kk4 & 16) ? 8 : 0);
      int ln = (r & 15) + 16 * kg;
      int mT = r >> 4;
      v4h hv;
      for (int q = 0; q < 4; ++q) hv[q] = (_Float16)vf[q];
      *(v4h*)&aS[buf][(mT * 32 + ln) * FR + e] = hv;
    }
  };
  // stage one 32x128 B tile, swizzled to fragment order
  auto stageB = [&](int buf, int k0) {
#pragma unroll
    for (int j = 0; j < 4; ++j) {
      int task = tid + 256 * j;       // 1024 quad-tasks
      int cc   = task & 127;
      int kk4  = (task >> 7) * 4;     // 0,4,...,28
      int gn = n0 + cc;
      v4h hv;
      for (int q = 0; q < 4; ++q) {
        int gk = k0 + kk4 + q;
        float vv = (gk < K && gn < N) ? W[(size_t)gk * N + gn] : 0.0f;
        hv[q] = (_Float16)vv;
      }
      int kg = (kk4 >> 3) & 1;
      int e  = (kk4 & 7) + ((kk4 & 16) ? 8 : 0);
      int ln = (cc & 15) + 16 * kg;
      int nT = cc >> 4;
      *(v4h*)&bS[buf][(nT * 32 + ln) * FR + e] = hv;
    }
  };

  stageA(0, 0);
  stageB(0, 0);
  int cur = 0;
  for (int k0 = 0; k0 < K; k0 += 32) {
    __syncthreads();
    if (k0 + 32 < K) {               // prefetch next tile into other buffer
      stageA(cur ^ 1, k0 + 32);
      stageB(cur ^ 1, k0 + 32);
    }
    // issue ALL fragment loads first (10x ds_load_b128), then the WMMA burst
    Frag fa;
    const _Float16* ap = &aS[cur][(waveM * 32 + lane) * FR];
    fa.h[0] = *(const v8h*)ap;
    fa.h[1] = *(const v8h*)(ap + 8);
    Frag fb[4];
#pragma unroll
    for (int nt = 0; nt < 4; ++nt) {
      const _Float16* bp16 = &bS[cur][((waveN * 4 + nt) * 32 + lane) * FR];
      fb[nt].h[0] = *(const v8h*)bp16;
      fb[nt].h[1] = *(const v8h*)(bp16 + 8);
    }
#pragma unroll
    for (int nt = 0; nt < 4; ++nt) {
      acc[nt] = __builtin_amdgcn_wmma_f32_16x16x32_f16(
          false, fa.v, false, fb[nt].v, (short)0, acc[nt], false, false);
    }
    cur ^= 1;
  }

  // epilogue: bias + activation + residual, fp32 store
  for (int nt = 0; nt < 4; ++nt) {
    int n = n0 + waveN * 64 + nt * 16 + row_l;
    if (n >= N) continue;
    float bv = bias ? bias[n] : 0.0f;
    for (int r = 0; r < 8; ++r) {
      int m = m0 + waveM * 16 + r + (kgrp ? 8 : 0);
      if (m >= M) continue;
      float v = acc[nt][r] + bv;
      if (act == 1) v = gelu_f(v);
      if (resid) v += resid[(size_t)m * N + n];
      C[(size_t)m * N + n] = v;
    }
  }
}

// ---------------------------------------------------------------------------
// LayerNorm over last dim, wave-per-row (wave32 shuffle reductions).
// Safe to call in-place (Y == X).
// ---------------------------------------------------------------------------
__global__ __launch_bounds__(256) void k_layernorm(
    const float* __restrict__ X, const float* __restrict__ g,
    const float* __restrict__ b, float* __restrict__ Y, int rows, int C)
{
  int wid = threadIdx.x >> 5, lane = threadIdx.x & 31;
  int row = blockIdx.x * 8 + wid;
  if (row >= rows) return;
  const float* x = X + (size_t)row * C;
  float s = 0.0f, ss = 0.0f;
  for (int c = lane; c < C; c += 32) { float v = x[c]; s += v; ss += v * v; }
  for (int o = 16; o > 0; o >>= 1) { s += __shfl_xor(s, o); ss += __shfl_xor(ss, o); }
  float m   = s / (float)C;
  float var = ss / (float)C - m * m;
  float inv = rsqrtf(var + 1e-5f);
  float* y = Y + (size_t)row * C;
  for (int c = lane; c < C; c += 32) y[c] = (x[c] - m) * inv * g[c] + b[c];
}

// ---------------------------------------------------------------------------
// Per-patch: row-normalize 16x4, Gram 4x4, Jacobi eigh, token/Vt/U outputs.
// ---------------------------------------------------------------------------
__global__ __launch_bounds__(256) void k_svd_token(
    const float* __restrict__ Mraw, float* __restrict__ tok,
    float* __restrict__ Vt, float* __restrict__ U, int BN)
{
  int p = blockIdx.x * blockDim.x + threadIdx.x;
  if (p >= BN) return;
  float Am[16][4];
  const float* src = Mraw + (size_t)p * 64;
  for (int r = 0; r < 16; ++r) {
    float n2 = 0.0f;
    for (int c = 0; c < 4; ++c) { float v = src[r * 4 + c]; Am[r][c] = v; n2 += v * v; }
    float n = fmaxf(sqrtf(n2), 1e-12f);
    for (int c = 0; c < 4; ++c) Am[r][c] /= n;
  }
  float G[4][4];
  for (int i = 0; i < 4; ++i)
    for (int j = 0; j < 4; ++j) {
      float s = 0.0f;
      for (int r = 0; r < 16; ++r) s += Am[r][i] * Am[r][j];
      G[i][j] = s;
    }
  float Vv[4][4] = {{1,0,0,0},{0,1,0,0},{0,0,1,0},{0,0,0,1}};
  for (int sweep = 0; sweep < 12; ++sweep) {
    for (int pp = 0; pp < 3; ++pp)
      for (int q = pp + 1; q < 4; ++q) {
        float apq = G[pp][q];
        if (fabsf(apq) < 1e-13f) continue;
        float tau = (G[q][q] - G[pp][pp]) / (2.0f * apq);
        float t   = (tau >= 0.0f ? 1.0f : -1.0f) / (fabsf(tau) + sqrtf(1.0f + tau * tau));
        float cth = rsqrtf(1.0f + t * t);
        float sth = t * cth;
        for (int kk = 0; kk < 4; ++kk) {
          float a = G[kk][pp], bq = G[kk][q];
          G[kk][pp] = cth * a - sth * bq;
          G[kk][q]  = sth * a + cth * bq;
        }
        for (int kk = 0; kk < 4; ++kk) {
          float a = G[pp][kk], bq = G[q][kk];
          G[pp][kk] = cth * a - sth * bq;
          G[q][kk]  = sth * a + cth * bq;
        }
        for (int kk = 0; kk < 4; ++kk) {
          float a = Vv[kk][pp], bq = Vv[kk][q];
          Vv[kk][pp] = cth * a - sth * bq;
          Vv[kk][q]  = sth * a + cth * bq;
        }
      }
  }
  float w[4]; for (int i = 0; i < 4; ++i) w[i] = G[i][i];
  int idx[4] = {0, 1, 2, 3};
  for (int i = 0; i < 3; ++i)
    for (int j = i + 1; j < 4; ++j)
      if (w[idx[j]] < w[idx[i]]) { int t2 = idx[i]; idx[i] = idx[j]; idx[j] = t2; }
  float s1 = 0, s2 = 0, s3 = 0, s4 = 0;
  for (int i = 0; i < 4; ++i) {
    float e = w[i]; s1 += e; float e2 = e * e; s2 += e2; s3 += e2 * e; s4 += e2 * e2;
  }
  float e1 = s1;
  float e2c = (e1 * s1 - s2) * 0.5f;
  float e3c = (e2c * s1 - e1 * s2 + s3) / 3.0f;
  float e4c = (e3c * s1 - e2c * s2 + e1 * s3 - s4) * 0.25f;
  float T[16];
  for (int i = 0; i < 4; ++i) T[i] = sqrtf(fmaxf(w[idx[3 - i]], 1e-24f));
  for (int i = 0; i < 4; ++i) {
    float fr = fmaxf(w[idx[i]], 0.0f);
    T[4 + i] = log1pf(fr);
    T[8 + i] = expf(-fr);
  }
  T[12] = e1; T[13] = e2c; T[14] = e3c; T[15] = e4c;
  for (int i = 0; i < 16; ++i) tok[(size_t)p * 16 + i] = T[i];
  for (int i = 0; i < 4; ++i) {
    int di = idx[3 - i];
    for (int j = 0; j < 4; ++j) Vt[(size_t)p * 16 + i * 4 + j] = Vv[j][di];
  }
  if (U) {
    for (int r = 0; r < 16; ++r)
      for (int j = 0; j < 4; ++j) {
        int dj = idx[3 - j];
        float s = 0.0f;
        for (int c = 0; c < 4; ++c) s += Am[r][c] * Vv[c][dj];
        U[(size_t)p * 64 + r * 4 + j] = s / fmaxf(T[j], 1e-16f);
      }
  }
}

// ---------------------------------------------------------------------------
// Tiny 4-token attention, one (group, head) per thread.
// ---------------------------------------------------------------------------
__global__ __launch_bounds__(256) void k_attn(
    const float* __restrict__ QKV, float* __restrict__ O,
    int groups, int C, int heads)
{
  int i = blockIdx.x * blockDim.x + threadIdx.x;
  if (i >= groups * heads) return;
  int g = i / heads, h = i % heads;
  int hd = C / heads;
  const float* base = QKV + (size_t)g * 4 * 3 * C;
  float s[4][4];
  float scale = rsqrtf((float)hd);
  for (int t = 0; t < 4; ++t)
    for (int u = 0; u < 4; ++u) {
      const float* q = base + (size_t)t * 3 * C + h * hd;
      const float* k = base + (size_t)u * 3 * C + C + h * hd;
      float a = 0.0f;
      for (int j = 0; j < hd; ++j) a += q[j] * k[j];
      s[t][u] = a * scale;
    }
  for (int t = 0; t < 4; ++t) {
    float mx = fmaxf(fmaxf(s[t][0], s[t][1]), fmaxf(s[t][2], s[t][3]));
    float den = 0.0f;
    for (int u = 0; u < 4; ++u) { s[t][u] = expf(s[t][u] - mx); den += s[t][u]; }
    float rr = 1.0f / den;
    for (int u = 0; u < 4; ++u) s[t][u] *= rr;
  }
  for (int t = 0; t < 4; ++t) {
    float* o = O + ((size_t)g * 4 + t) * C + h * hd;
    for (int j = 0; j < hd; ++j) {
      float a = 0.0f;
      for (int u = 0; u < 4; ++u) a += s[t][u] * base[(size_t)u * 3 * C + 2 * C + h * hd + j];
      o[j] = a;
    }
  }
}

// ------------------------------ data plumbing ------------------------------
__global__ void k_extract_patches(const float* __restrict__ img, float* __restrict__ P,
                                  int B, int C, int H, int W)
{
  int gh = H / 4, gw = W / 4, pd = C * 16;
  size_t total = (size_t)B * gh * gw * pd;
  size_t i = (size_t)blockIdx.x * blockDim.x + threadIdx.x;
  if (i >= total) return;
  int e = (int)(i % pd); size_t r = i / pd;
  int n = (int)(r % (gh * gw)); int b = (int)(r / (gh * gw));
  int c = e / 16, py = (e % 16) / 4, px = e % 4;
  int gy = n / gw, gx = n % gw;
  P[i] = img[(((size_t)b * C + c) * H + (gy * 4 + py)) * W + gx * 4 + px];
}

__global__ void k_stitch(const float* __restrict__ P, float* __restrict__ img,
                         int B, int H, int W)
{
  int gw = W / 4;
  size_t total = (size_t)B * 3 * H * W;
  size_t i = (size_t)blockIdx.x * blockDim.x + threadIdx.x;
  if (i >= total) return;
  int x = (int)(i % W); size_t r = i / W;
  int y = (int)(r % H); r /= H;
  int c = (int)(r % 3); int b = (int)(r / 3);
  int gy = y / 4, gx = x / 4, py = y % 4, px = x % 4;
  img[i] = P[((size_t)b * (H / 4) * gw + gy * gw + gx) * 48 + c * 16 + py * 4 + px];
}

__global__ void k_group2x2(const float* __restrict__ in, float* __restrict__ out,
                           int B, int gh, int gw, int C)
{
  size_t total = (size_t)B * gh * gw * C;
  size_t i = (size_t)blockIdx.x * blockDim.x + threadIdx.x;
  if (i >= total) return;
  int c = (int)(i % C); size_t r = i / C;
  int n = (int)(r % (gh * gw)); int b = (int)(r / (gh * gw));
  int gy = n / gw, gx = n % gw;
  int gy2 = gy >> 1, sy = gy & 1, gx2 = gx >> 1, sx = gx & 1;
  int ng = (gh / 2) * (gw / 2);
  out[(((size_t)b * ng + gy2 * (gw / 2) + gx2) * 4 + sy * 2 + sx) * C + c] = in[i];
}

__global__ void k_ungroup2x2(const float* __restrict__ in, float* __restrict__ out,
                             int B, int gh, int gw, int C)
{
  size_t total = (size_t)B * gh * gw * C;
  size_t i = (size_t)blockIdx.x * blockDim.x + threadIdx.x;
  if (i >= total) return;
  int c = (int)(i % C); size_t r = i / C;
  int n = (int)(r % (gh * gw)); int b = (int)(r / (gh * gw));
  int gy = n / gw, gx = n % gw;
  int gy2 = gy >> 1, sy = gy & 1, gx2 = gx >> 1, sx = gx & 1;
  int ng = (gh / 2) * (gw / 2);
  out[i] = in[(((size_t)b * ng + gy2 * (gw / 2) + gx2) * 4 + sy * 2 + sx) * C + c];
}

__global__ void k_fuse_dec(const float* __restrict__ expn, const float* __restrict__ tokS,
                           const float* __restrict__ VtS, float* __restrict__ out, int BN)
{
  size_t total = (size_t)BN * 4 * 160;
  size_t i = (size_t)blockIdx.x * blockDim.x + threadIdx.x;
  if (i >= total) return;
  int c = (int)(i % 160); size_t r = i / 160;
  int s = (int)(r & 3); size_t g = r >> 2;
  float v;
  if (c < 128)      v = expn[g * 512 + (size_t)s * 128 + c];
  else if (c < 144) v = tokS[g * 16 + (c - 128)];
  else              v = VtS[g * 16 + (c - 144)];
  out[i] = v;
}

__global__ void k_fuse_dec0(const float* __restrict__ tokens, const float* __restrict__ tok0,
                            const float* __restrict__ Vt0, const float* __restrict__ U0,
                            float* __restrict__ out, int BN)
{
  size_t total = (size_t)BN * 112;
  size_t i = (size_t)blockIdx.x * blockDim.x + threadIdx.x;
  if (i >= total) return;
  int c = (int)(i % 112); size_t r = i / 112;
  float v;
  if (c < 16)      v = tokens[r * 16 + c];
  else if (c < 32) v = tok0[r * 16 + (c - 16)];
  else if (c < 48) v = Vt0[r * 16 + (c - 32)];
  else             v = U0[r * 64 + (c - 48)];
  out[i] = v;
}

__global__ void k_conv3(const float* __restrict__ X, const float* __restrict__ Wc,
                        const float* __restrict__ bc, const float* __restrict__ resid,
                        float* __restrict__ Y, int B, int Ci, int Co, int H, int Wd, int act)
{
  size_t total = (size_t)B * Co * H * Wd;
  size_t i = (size_t)blockIdx.x * blockDim.x + threadIdx.x;
  if (i >= total) return;
  int x = (int)(i % Wd); size_t r = i / Wd;
  int y = (int)(r % H); r /= H;
  int oc = (int)(r % Co); int b = (int)(r / Co);
  float acc = bc[oc];
  for (int ic = 0; ic < Ci; ++ic)
    for (int ky = 0; ky < 3; ++ky) {
      int yy = y + ky - 1; if (yy < 0 || yy >= H) continue;
      for (int kx = 0; kx < 3; ++kx) {
        int xx = x + kx - 1; if (xx < 0 || xx >= Wd) continue;
        acc += X[(((size_t)b * Ci + ic) * H + yy) * Wd + xx] *
               Wc[((oc * Ci + ic) * 3 + ky) * 3 + kx];
      }
    }
  if (act) acc = gelu_f(acc);
  if (resid) acc += resid[i];
  Y[i] = acc;
}

// ------------------------------- host side ---------------------------------
struct Dense { const float* w; const float* b; };
struct Blk   { const float* ng; const float* nb; Dense l1, l2; };
struct GAttn { const float* ng; const float* nb; Dense qkv, out; };

struct Bump {
  char* base; size_t off;
  float* alloc(size_t nfloats) {
    size_t o = off;
    off += ((nfloats * sizeof(float) + 255) / 256) * 256;
    return (float*)(base + o);
  }
};

extern "C" void kernel_launch(void* const* d_in, const int* in_sizes, int n_in,
                              void* d_out, int out_size, void* d_ws, size_t ws_size,
                              hipStream_t stream)
{
  (void)in_sizes; (void)n_in; (void)out_size; (void)ws_size;
  const float* images = (const float*)d_in[0];
  int pi = 1;
  auto F  = [&]() -> const float* { return (const float*)d_in[pi++]; };
  auto Dn = [&]() { Dense d; d.w = F(); d.b = F(); return d; };
  auto Bk = [&]() { Blk b; b.ng = F(); b.nb = F(); b.l1 = Dn(); b.l2 = Dn(); return b; };
  auto Ga = [&]() { GAttn g; g.ng = F(); g.nb = F(); g.qkv = Dn(); g.out = Dn(); return g; };

  Dense e0_in = Dn(); Blk e0b[4]; for (int i = 0; i < 4; ++i) e0b[i] = Bk();
  Dense e0_out = Dn();
  GAttn ga1 = Ga(); Dense mi1 = Dn(); Blk bk1 = Bk(); Dense mo1 = Dn();
  GAttn ga2 = Ga(); Dense mi2 = Dn(); Blk bk2 = Bk(); Dense mo2 = Dn();
  Dense ex2 = Dn(); Dense in2 = Dn(); const float* ing2 = F(); const float* inb2 = F();
  GAttn gad2 = Ga(); const float* rng2 = F(); const float* rnb2 = F();
  Dense r1_2 = Dn(); Dense r2_2 = Dn();
  Dense ex1 = Dn(); Dense in1 = Dn(); const float* ing1 = F(); const float* inb1 = F();
  GAttn gad1 = Ga(); const float* rng1 = F(); const float* rnb1 = F();
  Dense r1_1 = Dn(); Dense r2_1 = Dn();
  Dense fu0 = Dn(); const float* fng = F(); const float* fnb = F();
  Blk d0b[4]; for (int i = 0; i < 4; ++i) d0b[i] = Bk();
  Dense out0 = Dn();
  const float* sw1 = F(); const float* sb1 = F();
  const float* sw2 = F(); const float* sb2 = F();

  const int B = 16;
  const int BN0 = B * 4096;
  const int BN1 = B * 1024;
  const int BN2 = B * 256;

  auto gemm = [&](const float* A, const Dense& d, const float* resid, float* Cc,
                  int M, int K, int N, int act) {
    dim3 grid((unsigned)((N + 127) / 128), (unsigned)((M + 63) / 64));
    k_gemm_wmma<<<grid, 256, 0, stream>>>(A, d.w, d.b, resid, Cc, M, K, N, act);
  };
  auto ln = [&](const float* X, const float* g, const float* b, float* Y, int rows, int C) {
    k_layernorm<<<(unsigned)((rows + 7) / 8), 256, 0, stream>>>(X, g, b, Y, rows, C);
  };
  auto blocks1d = [](size_t tot) { return (unsigned)((tot + 255) / 256); };

  Bump bp{(char*)d_ws, 0};
  float* patches = bp.alloc((size_t)BN0 * 48);
  float* tok0 = bp.alloc((size_t)BN0 * 16);
  float* Vt0  = bp.alloc((size_t)BN0 * 16);
  float* U0   = bp.alloc((size_t)BN0 * 64);
  float* tok1 = bp.alloc((size_t)BN1 * 16);
  float* Vt1  = bp.alloc((size_t)BN1 * 16);
  float* tok2 = bp.alloc((size_t)BN2 * 16);
  float* Vt2  = bp.alloc((size_t)BN2 * 16);
  float* D1b  = bp.alloc((size_t)B * 1024 * 16);
  float* D0b  = bp.alloc((size_t)B * 4096 * 16);
  float* imgb = bp.alloc((size_t)B * 3 * 256 * 256);
  float* convh = bp.alloc((size_t)B * 8 * 256 * 256);
  float* GRP2 = bp.alloc((size_t)BN2 * 4 * 16);
  float* GRP1 = bp.alloc((size_t)BN1 * 4 * 16);
  const size_t mark = bp.off;

  // ---- enc0: patch encoder ----
  {
    bp.off = mark;
    float* Hb = bp.alloc((size_t)BN0 * 256);
    float* Zb = bp.alloc((size_t)BN0 * 256);
    float* Tb = bp.alloc((size_t)BN0 * 256);
    float* Mb = bp.alloc((size_t)BN0 * 64);
    k_extract_patches<<<blocks1d((size_t)BN0 * 48), 256, 0, stream>>>(images, patches, B, 3, 256, 256);
    gemm(patches, e0_in, nullptr, Hb, BN0, 48, 256, 1);
    for (int i = 0; i < 4; ++i) {
      ln(Hb, e0b[i].ng, e0b[i].nb, Zb, BN0, 256);
      gemm(Zb, e0b[i].l1, nullptr, Tb, BN0, 256, 256, 1);
      gemm(Tb, e0b[i].l2, Hb, Hb, BN0, 256, 256, 0);
    }
    gemm(Hb, e0_out, nullptr, Mb, BN0, 256, 64, 0);
    k_svd_token<<<blocks1d(BN0), 256, 0, stream>>>(Mb, tok0, Vt0, U0, BN0);
  }

  // ---- encoder stages ----
  auto run_enc = [&](const float* tokIn, int ghF, int gwF, int BN,
                     GAttn& ga, Dense& mi, Blk& bk, Dense& mo,
                     float* tokOut, float* VtOut) {
    bp.off = mark;
    int rows = BN * 4;
    float* Gr  = bp.alloc((size_t)rows * 16);
    float* XN  = bp.alloc((size_t)rows * 16);
    float* QKV = bp.alloc((size_t)rows * 48);
    float* Ob  = bp.alloc((size_t)rows * 16);
    float* ATT = bp.alloc((size_t)rows * 16);
    float* Hb  = bp.alloc((size_t)BN * 128);
    float* Zb  = bp.alloc((size_t)BN * 128);
    float* Tb  = bp.alloc((size_t)BN * 128);
    float* Mb  = bp.alloc((size_t)BN * 64);
    k_group2x2<<<blocks1d((size_t)B * ghF * gwF * 16), 256, 0, stream>>>(tokIn, Gr, B, ghF, gwF, 16);
    ln(Gr, ga.ng, ga.nb, XN, rows, 16);
    gemm(XN, ga.qkv, nullptr, QKV, rows, 16, 48, 0);
    k_attn<<<blocks1d((size_t)BN * 2), 256, 0, stream>>>(QKV, Ob, BN, 16, 2);
    gemm(Ob, ga.out, Gr, ATT, rows, 16, 16, 0);
    gemm(ATT, mi, nullptr, Hb, BN, 64, 128, 1);
    ln(Hb, bk.ng, bk.nb, Zb, BN, 128);
    gemm(Zb, bk.l1, nullptr, Tb, BN, 128, 128, 1);
    gemm(Tb, bk.l2, Hb, Hb, BN, 128, 128, 0);
    gemm(Hb, mo, nullptr, Mb, BN, 128, 64, 0);
    k_svd_token<<<blocks1d(BN), 256, 0, stream>>>(Mb, tokOut, VtOut, nullptr, BN);
  };
  run_enc(tok0, 64, 64, BN1, ga1, mi1, bk1, mo1, tok1, Vt1);
  run_enc(tok1, 32, 32, BN2, ga2, mi2, bk2, mo2, tok2, Vt2);

  // ---- decoder stages ----
  auto run_dec = [&](const float* parent, int BN, const float* tokS, const float* VtS,
                     Dense& ex, Dense& inj, const float* ing, const float* inb,
                     GAttn& ga, const float* rng, const float* rnb,
                     Dense& r1, Dense& r2, float* outGrp) {
    bp.off = mark;
    int rows = BN * 4;
    float* EXP   = bp.alloc((size_t)BN * 512);
    float* FUSED = bp.alloc((size_t)rows * 160);
    float* FH    = bp.alloc((size_t)rows * 128);
    float* XN    = bp.alloc((size_t)rows * 128);
    float* QKV   = bp.alloc((size_t)rows * 384);
    float* Ob    = bp.alloc((size_t)rows * 128);
    float* ATT   = bp.alloc((size_t)rows * 128);
    gemm(parent, ex, nullptr, EXP, BN, 16, 512, 0);
    k_fuse_dec<<<blocks1d((size_t)rows * 160), 256, 0, stream>>>(EXP, tokS, VtS, FUSED, BN);
    gemm(FUSED, inj, nullptr, FH, rows, 160, 128, 1);
    ln(FH, ing, inb, FH, rows, 128);
    ln(FH, ga.ng, ga.nb, XN, rows, 128);
    gemm(XN, ga.qkv, nullptr, QKV, rows, 128, 384, 0);
    k_attn<<<blocks1d((size_t)BN * 2), 256, 0, stream>>>(QKV, Ob, BN, 128, 2);
    gemm(Ob, ga.out, FH, ATT, rows, 128, 128, 0);
    ln(ATT, rng, rnb, XN, rows, 128);
    gemm(XN, r1, nullptr, Ob, rows, 128, 128, 1);
    gemm(Ob, r2, nullptr, outGrp, rows, 128, 16, 0);
  };

  run_dec(tok2, BN2, tok2, Vt2, ex2, in2, ing2, inb2, gad2, rng2, rnb2, r1_2, r2_2, GRP2);
  k_ungroup2x2<<<blocks1d((size_t)B * 32 * 32 * 16), 256, 0, stream>>>(GRP2, D1b, B, 32, 32, 16);

  run_dec(D1b, BN1, tok1, Vt1, ex1, in1, ing1, inb1, gad1, rng1, rnb1, r1_1, r2_1, GRP1);
  k_ungroup2x2<<<blocks1d((size_t)B * 64 * 64 * 16), 256, 0, stream>>>(GRP1, D0b, B, 64, 64, 16);

  // ---- dec0: patch decoder + stitch + smoothing convs ----
  {
    bp.off = mark;
    float* FUSED0 = bp.alloc((size_t)BN0 * 112);
    float* Hb = bp.alloc((size_t)BN0 * 256);
    float* Zb = bp.alloc((size_t)BN0 * 256);
    float* Tb = bp.alloc((size_t)BN0 * 256);
    float* PO = bp.alloc((size_t)BN0 * 48);
    k_fuse_dec0<<<blocks1d((size_t)BN0 * 112), 256, 0, stream>>>(D0b, tok0, Vt0, U0, FUSED0, BN0);
    gemm(FUSED0, fu0, nullptr, Hb, BN0, 112, 256, 1);
    ln(Hb, fng, fnb, Hb, BN0, 256);
    for (int i = 0; i < 4; ++i) {
      ln(Hb, d0b[i].ng, d0b[i].nb, Zb, BN0, 256);
      gemm(Zb, d0b[i].l1, nullptr, Tb, BN0, 256, 256, 1);
      gemm(Tb, d0b[i].l2, Hb, Hb, BN0, 256, 256, 0);
    }
    gemm(Hb, out0, nullptr, PO, BN0, 256, 48, 0);
    k_stitch<<<blocks1d((size_t)B * 3 * 256 * 256), 256, 0, stream>>>(PO, imgb, B, 256, 256);
    k_conv3<<<blocks1d((size_t)B * 8 * 256 * 256), 256, 0, stream>>>(
        imgb, sw1, sb1, nullptr, convh, B, 3, 8, 256, 256, 1);
    k_conv3<<<blocks1d((size_t)B * 3 * 256 * 256), 256, 0, stream>>>(
        convh, sw2, sb2, imgb, (float*)d_out, B, 8, 3, 256, 256, 0);
  }
}
